// Bottom_up_90469191123501
// MI455X (gfx1250) — compile-verified, matrix operand order checked by reading, not run on hardware
//
#include <hip/hip_runtime.h>

#define C_IN   256
#define KTOT   2304      // 9 * 256, tap-major: k = t*256 + c
#define KSTEPS 72        // 2304 / 32
#define MT_PER_WAVE 4    // M-tiles (16 out-channels each) per wave

typedef __attribute__((ext_vector_type(16))) __bf16 v16bf;
typedef __attribute__((ext_vector_type(8)))  __bf16 v8bf;
typedef __attribute__((ext_vector_type(8)))  float  v8f;

// ---------------------------------------------------------------------------
// Pack fp32 conv weights [O=256][C=256][3][3] into bf16 hi/lo arrays in WMMA
// 16x16x32 A-fragment order with TAP-MAJOR K: k = t*256 + c.
//   elem index = ((mt*72 + ks)*32 + lane)*16 + j
//   maps to W[m = mt*16 + (lane&15)][c(k)*9 + t(k)],  k = ks*32 + kmap(lane,j)
//   kmap: khalf = (lane>>4)*8 ; j<8 -> khalf+j ; j>=8 -> 16+khalf+(j-8)
// ---------------------------------------------------------------------------
__global__ void pack_w_kernel(const float* __restrict__ w,
                              unsigned short* __restrict__ ahi,
                              unsigned short* __restrict__ alo)
{
    int gid = blockIdx.x * blockDim.x + threadIdx.x;
    const int total = 16 * KSTEPS * 32 * 16;
    if (gid >= total) return;
    int j    = gid & 15;
    int lane = (gid >> 4) & 31;
    int tmp  = gid >> 9;
    int ks   = tmp % KSTEPS;
    int mt   = tmp / KSTEPS;
    int m     = mt * 16 + (lane & 15);
    int khalf = (lane >> 4) * 8;
    int k     = ks * 32 + ((j < 8) ? (khalf + j) : (16 + khalf + (j - 8)));
    int t     = k >> 8;         // tap-major
    int c     = k & 255;
    float v   = w[m * KTOT + c * 9 + t];
    __bf16 h  = (__bf16)v;
    __bf16 l  = (__bf16)(v - (float)h);
    ((__bf16*)ahi)[gid] = h;
    ((__bf16*)alo)[gid] = l;
}

// ---------------------------------------------------------------------------
// NCHW -> NHWC fp32 with a zeroed guard row at pixel index HW
// out layout: [n][HW+1][256]
// ---------------------------------------------------------------------------
__global__ void nhwc_f32_kernel(const float* __restrict__ x,
                                float* __restrict__ o, int H)
{
    int gid = blockIdx.x * blockDim.x + threadIdx.x;
    int HW = H * H;
    int total = 2 * (HW + 1) * C_IN;
    if (gid >= total) return;
    int c   = gid % C_IN;
    int pix = (gid / C_IN) % (HW + 1);
    int n   = gid / (C_IN * (HW + 1));
    o[gid] = (pix < HW) ? x[((size_t)n * C_IN + c) * HW + pix] : 0.f;
}

// ---------------------------------------------------------------------------
// NCHW -> NHWC bf16 hi/lo split with zeroed guard row (for the dense level)
// ---------------------------------------------------------------------------
__global__ void nhwc_bf16_kernel(const float* __restrict__ x,
                                 unsigned short* __restrict__ bhi,
                                 unsigned short* __restrict__ blo, int H)
{
    int gid = blockIdx.x * blockDim.x + threadIdx.x;
    int HW = H * H;
    int total = 2 * (HW + 1) * C_IN;
    if (gid >= total) return;
    int c   = gid % C_IN;
    int pix = (gid / C_IN) % (HW + 1);
    int n   = gid / (C_IN * (HW + 1));
    float v = (pix < HW) ? x[((size_t)n * C_IN + c) * HW + pix] : 0.f;
    __bf16 h = (__bf16)v;
    ((__bf16*)bhi)[gid] = h;
    ((__bf16*)blo)[gid] = (__bf16)(v - (float)h);
}

// ---------------------------------------------------------------------------
// bum = x + maxpool2x2(prev)   (prev is 2H x 2H, NCHW)
// ---------------------------------------------------------------------------
__global__ void bum_kernel(const float* __restrict__ x, const float* __restrict__ prev,
                           float* __restrict__ out, int H)
{
    int gid = blockIdx.x * blockDim.x + threadIdx.x;
    int HW = H * H;
    int total = 2 * C_IN * HW;
    if (gid >= total) return;
    int wq = gid % H;
    int h  = (gid / H) % H;
    int nc = gid / HW;
    int Hp = H * 2;
    const float* p  = prev + (size_t)nc * Hp * Hp;
    const float* r0 = p + (2 * h) * Hp + 2 * wq;
    const float* r1 = r0 + Hp;
    float m = fmaxf(fmaxf(r0[0], r0[1]), fmaxf(r1[0], r1[1]));
    out[gid] = x[gid] + m;
}

// ---------------------------------------------------------------------------
// Small 18-out-channel offset conv (direct fp32 VALU; <2% of total FLOPs)
// ---------------------------------------------------------------------------
__global__ void off_conv_kernel(const float* __restrict__ bum,
                                const float* __restrict__ ow,
                                const float* __restrict__ ob,
                                float* __restrict__ out, int H)
{
    int gid = blockIdx.x * blockDim.x + threadIdx.x;
    int HW = H * H;
    int total = 2 * 18 * HW;
    if (gid >= total) return;
    int p  = gid % HW;
    int oc = (gid / HW) % 18;
    int n  = gid / (HW * 18);
    int h = p / H, wq = p % H;
    const float* src = bum + (size_t)n * C_IN * HW;
    float acc = ob[oc];
    for (int t = 0; t < 9; ++t) {
        int yy = h + t / 3 - 1;
        int xx = wq + t % 3 - 1;
        if (yy < 0 || yy >= H || xx < 0 || xx >= H) continue;
        int base = yy * H + xx;
        const float* wp = ow + (size_t)oc * KTOT + t;
        for (int c = 0; c < C_IN; ++c)
            acc = fmaf(wp[c * 9], src[(size_t)c * HW + base], acc);
    }
    out[gid] = acc;
}

// ---------------------------------------------------------------------------
// Precompute bilinear taps: 4 indices + 4 weights per (n, tap, pixel);
// OOB folded into weight=0 with idx forced to 0 so gathers are unconditional.
// ---------------------------------------------------------------------------
__global__ void bilin_kernel(const float* __restrict__ offs,
                             int4* __restrict__ idx4, float4* __restrict__ wt4,
                             int H)
{
    int gid = blockIdx.x * blockDim.x + threadIdx.x;
    int HW = H * H;
    int total = 2 * 9 * HW;
    if (gid >= total) return;
    int p = gid % HW;
    int t = (gid / HW) % 9;
    int n = gid / (HW * 9);
    int h = p / H, wq = p % H;
    float dy = offs[((size_t)(n * 18 + t * 2 + 0)) * HW + p];
    float dx = offs[((size_t)(n * 18 + t * 2 + 1)) * HW + p];
    float py = (float)(h - 1 + t / 3) + dy;
    float px = (float)(wq - 1 + t % 3) + dx;
    float y0f = floorf(py), x0f = floorf(px);
    float fy = py - y0f, fx = px - x0f;
    int y0 = (int)y0f, x0 = (int)x0f;

    int   ii[4];
    float ww[4];
    int   ys[4] = { y0, y0,     y0 + 1, y0 + 1 };
    int   xs[4] = { x0, x0 + 1, x0,     x0 + 1 };
    float ws[4] = { (1.f - fy) * (1.f - fx), (1.f - fy) * fx,
                    fy * (1.f - fx),         fy * fx };
#pragma unroll
    for (int q = 0; q < 4; ++q) {
        bool v = (ys[q] >= 0) & (ys[q] < H) & (xs[q] >= 0) & (xs[q] < H);
        ii[q] = v ? (ys[q] * H + xs[q]) : 0;
        ww[q] = v ? ws[q] : 0.f;
    }
    idx4[gid] = make_int4(ii[0], ii[1], ii[2], ii[3]);
    wt4[gid]  = make_float4(ww[0], ww[1], ww[2], ww[3]);
}

// ---------------------------------------------------------------------------
// Dense 3x3 conv as implicit GEMM on the bf16 WMMA pipe.
// fp32 accuracy via hi/lo bf16 split: hi*hi + hi*lo + lo*hi.
// Tap-major K + NHWC bf16 activations: each B half-fragment is ONE contiguous
// 16B load (8 channels at one tap); OOB taps hit the zeroed guard row.
// One wave computes 64(M) x 16(N): 12 WMMAs per K-step against 4 B loads.
// ---------------------------------------------------------------------------
__global__ void __launch_bounds__(32)
conv_gemm_kernel(const unsigned short* __restrict__ bhi_u,
                 const unsigned short* __restrict__ blo_u,
                 const unsigned short* __restrict__ ahi_u,
                 const unsigned short* __restrict__ alo_u,
                 const float* __restrict__ bias,
                 float* __restrict__ out, int H)
{
    __shared__ int s_pix[16 * 9];   // guarded sample pixel per (pn, tap)

    const int lane  = threadIdx.x;
    const int HW    = H * H;
    const int ptile = blockIdx.x;
    const int mtb   = blockIdx.y * MT_PER_WAVE;
    const int n     = blockIdx.z;
    const int p0    = ptile * 16;

    for (int e = lane; e < 144; e += 32) {
        int pn = e / 9, t = e % 9;
        int p = p0 + pn;
        int h = p / H, wq = p % H;
        int yy = h + t / 3 - 1;
        int xx = wq + t % 3 - 1;
        bool v = (yy >= 0 && yy < H && xx >= 0 && xx < H);
        s_pix[e] = v ? (yy * H + xx) : HW;   // HW = zero guard row
    }
    __syncthreads();

    const __bf16* bhi = (const __bf16*)bhi_u + (size_t)n * (HW + 1) * C_IN;
    const __bf16* blo = (const __bf16*)blo_u + (size_t)n * (HW + 1) * C_IN;
    const int pn    = lane & 15;
    const int khalf = (lane >> 4) * 8;

    const v16bf* Ah[MT_PER_WAVE];
    const v16bf* Al[MT_PER_WAVE];
#pragma unroll
    for (int u = 0; u < MT_PER_WAVE; ++u) {
        Ah[u] = (const v16bf*)ahi_u + (size_t)(mtb + u) * KSTEPS * 32 + lane;
        Al[u] = (const v16bf*)alo_u + (size_t)(mtb + u) * KSTEPS * 32 + lane;
    }

    v8f acc[MT_PER_WAVE] = {};
    for (int ks = 0; ks < KSTEPS; ++ks) {
        int k1 = ks * 32 + khalf;          // run of 8: tap t1, channels c1..c1+7
        int t1 = k1 >> 8, c1 = k1 & 255;
        int k2 = k1 + 16;
        int t2 = k2 >> 8, c2 = k2 & 255;
        int px1 = s_pix[pn * 9 + t1];
        int px2 = s_pix[pn * 9 + t2];

        v8bf h0 = *(const v8bf*)(bhi + (size_t)px1 * C_IN + c1);
        v8bf h1 = *(const v8bf*)(bhi + (size_t)px2 * C_IN + c2);
        v8bf l0 = *(const v8bf*)(blo + (size_t)px1 * C_IN + c1);
        v8bf l1 = *(const v8bf*)(blo + (size_t)px2 * C_IN + c2);
        v16bf b_h = __builtin_shufflevector(h0, h1, 0,1,2,3,4,5,6,7,8,9,10,11,12,13,14,15);
        v16bf b_l = __builtin_shufflevector(l0, l1, 0,1,2,3,4,5,6,7,8,9,10,11,12,13,14,15);

#pragma unroll
        for (int u = 0; u < MT_PER_WAVE; ++u) {
            v16bf a_h = Ah[u][ks * 32];
            v16bf a_l = Al[u][ks * 32];
            acc[u] = __builtin_amdgcn_wmma_f32_16x16x32_bf16(false, a_h, false, b_h,
                                                             (short)0, acc[u], false, false);
            acc[u] = __builtin_amdgcn_wmma_f32_16x16x32_bf16(false, a_h, false, b_l,
                                                             (short)0, acc[u], false, false);
            acc[u] = __builtin_amdgcn_wmma_f32_16x16x32_bf16(false, a_l, false, b_h,
                                                             (short)0, acc[u], false, false);
        }
    }

    const int mhi = (lane >> 4) * 8;
#pragma unroll
    for (int u = 0; u < MT_PER_WAVE; ++u) {
#pragma unroll
        for (int r = 0; r < 8; ++r) {
            int m = (mtb + u) * 16 + r + mhi;
            out[((size_t)n * C_IN + m) * HW + p0 + pn] = acc[u][r] + bias[m];
        }
    }
}

// ---------------------------------------------------------------------------
// Deformable 3x3 conv as implicit GEMM: same skeleton; a B half-fragment is a
// 4-row bilinear blend of contiguous 8-channel fp32 NHWC vectors, then hi/lo
// bf16 split. Indices always safe (weights carry validity).
// ---------------------------------------------------------------------------
__global__ void __launch_bounds__(32)
dconv_gemm_kernel(const float* __restrict__ nhwc,
                  const int4* __restrict__ idx4,
                  const float4* __restrict__ wt4,
                  const unsigned short* __restrict__ ahi_u,
                  const unsigned short* __restrict__ alo_u,
                  const float* __restrict__ bias,
                  float* __restrict__ out, int H)
{
    __shared__ int4   s_idx[16 * 9];
    __shared__ float4 s_wt [16 * 9];

    const int lane  = threadIdx.x;
    const int HW    = H * H;
    const int ptile = blockIdx.x;
    const int mtb   = blockIdx.y * MT_PER_WAVE;
    const int n     = blockIdx.z;
    const int p0    = ptile * 16;

    for (int e = lane; e < 144; e += 32) {
        int pn = e / 9, t = e % 9;
        int p = p0 + pn;
        size_t g = (size_t)(n * 9 + t) * HW + p;
        s_idx[e] = idx4[g];
        s_wt[e]  = wt4[g];
    }
    __syncthreads();

    const float* src = nhwc + (size_t)n * (HW + 1) * C_IN;
    const int pn    = lane & 15;
    const int khalf = (lane >> 4) * 8;

    const v16bf* Ah[MT_PER_WAVE];
    const v16bf* Al[MT_PER_WAVE];
#pragma unroll
    for (int u = 0; u < MT_PER_WAVE; ++u) {
        Ah[u] = (const v16bf*)ahi_u + (size_t)(mtb + u) * KSTEPS * 32 + lane;
        Al[u] = (const v16bf*)alo_u + (size_t)(mtb + u) * KSTEPS * 32 + lane;
    }

    v8f acc[MT_PER_WAVE] = {};
    for (int ks = 0; ks < KSTEPS; ++ks) {
        v16bf b_h, b_l;
#pragma unroll
        for (int r = 0; r < 2; ++r) {
            int k  = ks * 32 + khalf + 16 * r;   // run of 8: one tap, 8 channels
            int t  = k >> 8;
            int c0 = k & 255;
            int4   ii = s_idx[pn * 9 + t];
            float4 ww = s_wt [pn * 9 + t];
            const float* base = src + c0;
            v8f p0v = *(const v8f*)(base + (size_t)ii.x * C_IN);
            v8f p1v = *(const v8f*)(base + (size_t)ii.y * C_IN);
            v8f p2v = *(const v8f*)(base + (size_t)ii.z * C_IN);
            v8f p3v = *(const v8f*)(base + (size_t)ii.w * C_IN);
            v8f v = p0v * ww.x + p1v * ww.y + p2v * ww.z + p3v * ww.w;
#pragma unroll
            for (int e = 0; e < 8; ++e) {
                __bf16 hv = (__bf16)v[e];
                b_h[r * 8 + e] = hv;
                b_l[r * 8 + e] = (__bf16)(v[e] - (float)hv);
            }
        }

#pragma unroll
        for (int u = 0; u < MT_PER_WAVE; ++u) {
            v16bf a_h = Ah[u][ks * 32];
            v16bf a_l = Al[u][ks * 32];
            acc[u] = __builtin_amdgcn_wmma_f32_16x16x32_bf16(false, a_h, false, b_h,
                                                             (short)0, acc[u], false, false);
            acc[u] = __builtin_amdgcn_wmma_f32_16x16x32_bf16(false, a_h, false, b_l,
                                                             (short)0, acc[u], false, false);
            acc[u] = __builtin_amdgcn_wmma_f32_16x16x32_bf16(false, a_l, false, b_h,
                                                             (short)0, acc[u], false, false);
        }
    }

    const int mhi = (lane >> 4) * 8;
#pragma unroll
    for (int u = 0; u < MT_PER_WAVE; ++u) {
#pragma unroll
        for (int r = 0; r < 8; ++r) {
            int m = (mtb + u) * 16 + r + mhi;
            out[((size_t)n * C_IN + m) * HW + p0 + pn] = acc[u][r] + bias[m];
        }
    }
}

// ---------------------------------------------------------------------------
extern "C" void kernel_launch(void* const* d_in, const int* in_sizes, int n_in,
                              void* d_out, int out_size, void* d_ws, size_t ws_size,
                              hipStream_t stream)
{
    (void)in_sizes; (void)n_in; (void)out_size; (void)ws_size;

    // setup_inputs() dict order:
    // 0..3: x0..x3 ; 4..11: w0,b0,w1,b1,w2,b2,w3,b3 ; 12..17: ow1,ob1,ow2,ob2,ow3,ob3
    const float* x[4]  = { (const float*)d_in[0], (const float*)d_in[1],
                           (const float*)d_in[2], (const float*)d_in[3] };
    const float* wt[4] = { (const float*)d_in[4], (const float*)d_in[6],
                           (const float*)d_in[8], (const float*)d_in[10] };
    const float* bs[4] = { (const float*)d_in[5], (const float*)d_in[7],
                           (const float*)d_in[9], (const float*)d_in[11] };
    const float* ow[3] = { (const float*)d_in[12], (const float*)d_in[14],
                           (const float*)d_in[16] };
    const float* ob[3] = { (const float*)d_in[13], (const float*)d_in[15],
                           (const float*)d_in[17] };
    float* out = (float*)d_out;

    const int Hs[4] = { 128, 64, 32, 16 };
    const size_t outOff[4] = { 0,
                               (size_t)2 * 256 * 128 * 128,
                               (size_t)2 * 256 * 128 * 128 + (size_t)2 * 256 * 64 * 64,
                               (size_t)2 * 256 * 128 * 128 + (size_t)2 * 256 * 64 * 64
                                 + (size_t)2 * 256 * 32 * 32 };

    // ---- workspace carve (~63 MB) ----
    const size_t PACKELEMS = (size_t)16 * KSTEPS * 32 * 16;     // 589824 per level
    const size_t NHWC0     = (size_t)2 * (128 * 128 + 1) * C_IN; // level-0 NHWC elems
    const size_t NHWC1     = (size_t)2 * (64 * 64 + 1) * C_IN;   // level-1 NHWC elems (max deform)
    char* ws = (char*)d_ws;
    unsigned short* Ahi = (unsigned short*)ws;  ws += 4 * PACKELEMS * sizeof(unsigned short);
    unsigned short* Alo = (unsigned short*)ws;  ws += 4 * PACKELEMS * sizeof(unsigned short);
    unsigned short* Bhi0 = (unsigned short*)ws; ws += NHWC0 * sizeof(unsigned short);
    unsigned short* Blo0 = (unsigned short*)ws; ws += NHWC0 * sizeof(unsigned short);
    float* bum   = (float*)ws;                  ws += (size_t)2 * C_IN * 64 * 64 * sizeof(float);
    float* nhwcF = (float*)ws;                  ws += NHWC1 * sizeof(float);
    float* offs  = (float*)ws;                  ws += (size_t)2 * 18 * 64 * 64 * sizeof(float);
    int4*   i4   = (int4*)ws;                   ws += (size_t)2 * 9 * 64 * 64 * sizeof(int4);
    float4* w4   = (float4*)ws;

    // ---- pack all four levels' weights into WMMA fragment order ----
    {
        int blocks = (int)((PACKELEMS + 255) / 256);
        for (int l = 0; l < 4; ++l)
            pack_w_kernel<<<blocks, 256, 0, stream>>>(wt[l],
                                                      Ahi + l * PACKELEMS,
                                                      Alo + l * PACKELEMS);
    }

    // ---- level 0: NHWC bf16 split + dense conv ----
    {
        int H = Hs[0], HW = H * H;
        int tot = (int)NHWC0;
        nhwc_bf16_kernel<<<(tot + 255) / 256, 256, 0, stream>>>(x[0], Bhi0, Blo0, H);
        dim3 grid(HW / 16, 16 / MT_PER_WAVE, 2);
        conv_gemm_kernel<<<grid, 32, 0, stream>>>(Bhi0, Blo0, Ahi, Alo, bs[0],
                                                  out + outOff[0], H);
    }

    // ---- levels 1..3: bum -> offset conv -> bilinear tables -> NHWC -> GEMM --
    for (int i = 1; i <= 3; ++i) {
        int H = Hs[i], HW = H * H;

        int tot = 2 * C_IN * HW;
        bum_kernel<<<(tot + 255) / 256, 256, 0, stream>>>(x[i], out + outOff[i - 1],
                                                          bum, H);

        int tot2 = 2 * 18 * HW;
        off_conv_kernel<<<(tot2 + 255) / 256, 256, 0, stream>>>(bum, ow[i - 1],
                                                                ob[i - 1], offs, H);

        int tot3 = 2 * 9 * HW;
        bilin_kernel<<<(tot3 + 255) / 256, 256, 0, stream>>>(offs, i4, w4, H);

        int tot4 = 2 * (HW + 1) * C_IN;
        nhwc_f32_kernel<<<(tot4 + 255) / 256, 256, 0, stream>>>(bum, nhwcF, H);

        dim3 grid(HW / 16, 16 / MT_PER_WAVE, 2);
        dconv_gemm_kernel<<<grid, 32, 0, stream>>>(nhwcF, i4, w4,
                                                   Ahi + i * PACKELEMS,
                                                   Alo + i * PACKELEMS,
                                                   bs[i], out + outOff[i], H);
    }
}
